// ExportCausalSelfAttention_8100308320384
// MI455X (gfx1250) — compile-verified, matrix operand order checked by reading, not run on hardware
//
#include <hip/hip_runtime.h>

// ---------------- problem constants ----------------
#define N_EMBD  2048
#define N_HEADS 16
#define N_KV    4
#define HD      128
#define T_SEQ   2048
#define QG_N    (N_HEADS * HD * 2)   // 4096
#define KV_N    (N_KV * HD)          // 512

// ---------------- CDNA5 WMMA types ----------------
typedef __attribute__((ext_vector_type(16))) __bf16       bf16x16;
typedef __attribute__((ext_vector_type(8)))  float        f32x8;
typedef __attribute__((ext_vector_type(4)))  unsigned int u32x4;
typedef __attribute__((ext_vector_type(8)))  unsigned int u32x8;

union Frag { bf16x16 v; u32x4 u[2]; };

__device__ __forceinline__ unsigned short f2bf(float f) {
  unsigned int u = __float_as_uint(f);
  u += 0x7FFFu + ((u >> 16) & 1u);      // round-to-nearest-even
  return (unsigned short)(u >> 16);
}

// load a 16x32 bf16 A/B-operand fragment from a row-major [rows x K] matrix in
// GLOBAL memory. ISA layout: lane L -> row (base + L%16); K-chunks
// [half*8..+7] and [16+half*8..+7], half = L/16. Two 16-byte loads per lane.
__device__ __forceinline__ Frag load_frag(const unsigned short* __restrict__ rowptr,
                                          int k0, int half) {
  Frag f;
  f.u[0] = *(const u32x4*)(rowptr + k0 + half * 8);
  f.u[1] = *(const u32x4*)(rowptr + k0 + 16 + half * 8);
  return f;
}

// Same fragment shape but from an LDS tile staged by the TDM with padding:
// row = 32 bf16 (64B data) + 16B pad -> row stride 40 ushorts (80B, 16B-aligned).
#define LDS_ROW 40
__device__ __forceinline__ Frag lds_frag(const unsigned short* base, int row, int half) {
  const unsigned short* r = base + row * LDS_ROW;
  Frag f;
  f.u[0] = *(const u32x4*)(r + half * 8);
  f.u[1] = *(const u32x4*)(r + 16 + half * 8);
  return f;
}

// ---------------- Tensor Data Mover: 2D bf16 tile -> LDS ----------------
// D# per CDNA5 ISA. data_size=2B; LDS padding enabled:
// pad_interval code 3 (every 16 DWORDs = one 64B row), pad_amount code 3
// (4 DWORDs = 16B) => LDS row stride 80B. Tracked on TENSORcnt.
__device__ __forceinline__ void tdm_load_2d_bf16(void* lds_dst, const void* gsrc,
                                                 unsigned tile_rows,
                                                 unsigned tile_cols,
                                                 unsigned long long row_stride_elems) {
  const unsigned long long ga = (unsigned long long)gsrc;
  const unsigned lds_addr = (unsigned)(unsigned long long)lds_dst; // flat[31:0] == LDS offset
  u32x4 g0;
  g0[0] = 1u;                                                   // count=1
  g0[1] = lds_addr;
  g0[2] = (unsigned)ga;
  g0[3] = ((unsigned)(ga >> 32) & 0x01FFFFFFu) | 0x80000000u;   // addr[56:32] | type=2
  const unsigned td0 = 0x7FFFFFFFu, td1 = 0x7FFFFFFFu;          // no OOB clipping
  u32x8 g1;
  g1[0] = (1u << 16) | (1u << 20) | (3u << 22) | (3u << 25);    // 2B, pad 16B/row
  g1[1] = (td0 & 0xFFFFu) << 16;                                // tensor_dim0 lo
  g1[2] = (td0 >> 16) | ((td1 & 0xFFFFu) << 16);                // dim0 hi | dim1 lo
  g1[3] = (td1 >> 16) | (tile_cols << 16);                      // dim1 hi | tile_dim0
  g1[4] = tile_rows & 0xFFFFu;                                  // tile_dim1 (tile_dim2=0)
  g1[5] = (unsigned)row_stride_elems;                           // dim0 stride lo
  g1[6] = (unsigned)(row_stride_elems >> 32) & 0xFFFFu;         // stride hi | dim1 stride=0
  g1[7] = 0u;
  asm volatile("tensor_load_to_lds %0, %1" :: "s"(g0), "s"(g1) : "memory");
}

// ---------------- fp32 -> bf16 convert ----------------
__global__ __launch_bounds__(256) void cvt_f32_bf16(const float* __restrict__ src,
                                                    unsigned short* __restrict__ dst,
                                                    long n) {
  long i = ((long)blockIdx.x * blockDim.x + threadIdx.x) * 4;
  if (i + 3 < n) {
    dst[i + 0] = f2bf(src[i + 0]);
    dst[i + 1] = f2bf(src[i + 1]);
    dst[i + 2] = f2bf(src[i + 2]);
    dst[i + 3] = f2bf(src[i + 3]);
  }
}

// ---------------- WMMA GEMM with TDM-staged LDS tiles ----------------
// C[MxN] = A[MxK] * B[NxK]^T. Block = 8 waves, 128x128 tile; wave (mg,ng)
// owns a 32x64 subtile (2x4 accumulators). Per k-step (K=32) wave 0 TDM-loads
// the 128x32 A and B tiles into double-buffered LDS; all waves ds_load_b128
// fragments. M,N multiples of 128; K multiple of 32.
__global__ __launch_bounds__(256) void gemm_bf16_nt(const unsigned short* __restrict__ A,
                                                    const unsigned short* __restrict__ B,
                                                    float* __restrict__ C,
                                                    int M, int N, int K) {
  __shared__ unsigned short Al[2][128 * LDS_ROW];
  __shared__ unsigned short Bl[2][128 * LDS_ROW];

  const int lane = threadIdx.x & 31;
  const int wave = threadIdx.x >> 5;
  const int half = lane >> 4;
  const int l16  = lane & 15;
  const int mg = wave >> 1;                 // 0..3 : 32-row M subtile
  const int ng = wave & 1;                  // 0..1 : 64-col N subtile
  const int m0 = blockIdx.y * 128;
  const int n0 = blockIdx.x * 128;

  f32x8 acc[2][4];
#pragma unroll
  for (int i = 0; i < 2; ++i)
#pragma unroll
    for (int j = 0; j < 4; ++j)
#pragma unroll
      for (int r = 0; r < 8; ++r) acc[i][j][r] = 0.0f;

  const unsigned short* arow = A + (size_t)m0 * K;
  const unsigned short* brow = B + (size_t)n0 * K;
  const int niter = K / 32;

  if (wave == 0) {          // TDM ignores EXEC; issue once per workgroup
    tdm_load_2d_bf16(&Al[0][0], arow + 0,  128, 32, (unsigned long long)K);
    tdm_load_2d_bf16(&Bl[0][0], brow + 0,  128, 32, (unsigned long long)K);
    tdm_load_2d_bf16(&Al[1][0], arow + 32, 128, 32, (unsigned long long)K);
    tdm_load_2d_bf16(&Bl[1][0], brow + 32, 128, 32, (unsigned long long)K);
  }

  for (int it = 0; it < niter; ++it) {
    if (wave == 0) __builtin_amdgcn_s_wait_tensorcnt(2); // stage `it` landed
    __syncthreads();

    const unsigned short* a = &Al[it & 1][0];
    const unsigned short* b = &Bl[it & 1][0];
    Frag af[2], bf[4];
#pragma unroll
    for (int i = 0; i < 2; ++i) af[i] = lds_frag(a, mg * 32 + i * 16 + l16, half);
#pragma unroll
    for (int j = 0; j < 4; ++j) bf[j] = lds_frag(b, ng * 64 + j * 16 + l16, half);
#pragma unroll
    for (int i = 0; i < 2; ++i)
#pragma unroll
      for (int j = 0; j < 4; ++j)
        acc[i][j] = __builtin_amdgcn_wmma_f32_16x16x32_bf16(
            false, af[i].v, false, bf[j].v, (short)0, acc[i][j], false, false);

    __syncthreads();                         // all waves done reading buffer
    const int knext = (it + 2) * 32;
    if (wave == 0 && knext < K) {            // refill the buffer just consumed
      tdm_load_2d_bf16(&Al[it & 1][0], arow + knext, 128, 32, (unsigned long long)K);
      tdm_load_2d_bf16(&Bl[it & 1][0], brow + knext, 128, 32, (unsigned long long)K);
    }
  }

  // C/D layout: element (vgpr r, lane) = C[row + r + 8*half][col + l16]
#pragma unroll
  for (int i = 0; i < 2; ++i)
#pragma unroll
    for (int j = 0; j < 4; ++j)
#pragma unroll
      for (int r = 0; r < 8; ++r)
        C[(size_t)(m0 + mg * 32 + i * 16 + r + 8 * half) * N +
          n0 + ng * 64 + j * 16 + l16] = acc[i][j][r];
}

// ---------------- fused RMSNorm + partial RoPE, fp32 -> bf16 ----------------
// one wave per (t, head); lane owns dims {lane, lane+32, lane+64, lane+96}.
// dst layout: [head][T][HD] bf16.
__global__ __launch_bounds__(256) void qk_norm_rope(const float* __restrict__ src,
                                                    int rowStride, int headStride,
                                                    const float* __restrict__ w,
                                                    const int* __restrict__ pos,
                                                    unsigned short* __restrict__ dst,
                                                    int T, int nheads) {
  const int lane = threadIdx.x & 31;
  const int wid = blockIdx.x * (blockDim.x >> 5) + (threadIdx.x >> 5);
  if (wid >= T * nheads) return;
  const int t = wid / nheads;
  const int h = wid - t * nheads;

  const float* base = src + (size_t)t * rowStride + (size_t)h * headStride;
  float x0 = base[lane];
  float x1 = base[lane + 32];
  float x2 = base[lane + 64];
  float x3 = base[lane + 96];

  float ss = x0 * x0 + x1 * x1 + x2 * x2 + x3 * x3;
#pragma unroll
  for (int m = 1; m < 32; m <<= 1) ss += __shfl_xor(ss, m, 32);
  const float inv = rsqrtf(ss * (1.0f / 128.0f) + 1e-6f);

  x0 *= inv * w[lane];
  x1 *= inv * w[lane + 32];
  x2 *= inv * w[lane + 64];
  x3 *= inv * w[lane + 96];

  // partial rope: rot_dim = 64, pairs (i, i+32), theta = 1e6
  const float p = (float)pos[t];
  const float f = p * __powf(1.0e6f, -(float)lane * (1.0f / 32.0f));
  float sn, cs;
  __sincosf(f, &sn, &cs);
  const float r0 = x0 * cs - x1 * sn;
  const float r1 = x1 * cs + x0 * sn;

  unsigned short* out = dst + ((size_t)h * T + t) * HD;
  out[lane]      = f2bf(r0);
  out[lane + 32] = f2bf(r1);
  out[lane + 64] = f2bf(x2);
  out[lane + 96] = f2bf(x3);
}

// ---------------- V transpose: (T, KV*HD) f32 -> (KV*HD, T) bf16 ----------------
__global__ __launch_bounds__(256) void v_transpose(const float* __restrict__ Vp,
                                                   unsigned short* __restrict__ Vt,
                                                   int T) {
  const int idx = blockIdx.x * blockDim.x + threadIdx.x;   // over KV_N * T
  if (idx >= KV_N * T) return;
  const int c = idx >> 11;          // T == 2048
  const int t = idx & (T_SEQ - 1);
  Vt[idx] = f2bf(Vp[(size_t)t * KV_N + c]);               // coalesced write
}

// ---------------- flash attention: one wave per (head, 16-query tile) --------
// Qb: [H][T][HD] bf16 ; Kb: [KV][T][HD] bf16 ; Vt: [KV][HD][T] bf16
// Y : [T][H*HD] f32
__global__ __launch_bounds__(32) void flash_attn(const unsigned short* __restrict__ Qb,
                                                 const unsigned short* __restrict__ Kb,
                                                 const unsigned short* __restrict__ Vt,
                                                 float* __restrict__ Y, int T) {
  const int h  = blockIdx.y;
  const int m0 = blockIdx.x * 16;
  const int kv = h >> 2;                       // G = 4
  const int lane = threadIdx.x & 31;
  const int half = lane >> 4;
  const int l16  = lane & 15;

  __shared__ unsigned short Pl[16 * 32];       // P staged for A-operand reshape

  // Q fragments for the 4 K-steps over HD=128
  Frag qf[4];
  const unsigned short* qrow = Qb + ((size_t)h * T + m0 + l16) * HD;
#pragma unroll
  for (int j = 0; j < 4; ++j) qf[j] = load_frag(qrow, j * 32, half);

  f32x8 o[8];
#pragma unroll
  for (int db = 0; db < 8; ++db)
#pragma unroll
    for (int r = 0; r < 8; ++r) o[db][r] = 0.0f;

  float mrow[8], lrow[8];
#pragma unroll
  for (int r = 0; r < 8; ++r) { mrow[r] = -1e30f; lrow[r] = 0.0f; }

  const float scale = 0.08838834764831845f;    // 1/sqrt(128)
  const int nkb = (m0 + 16 + 31) >> 5;         // causal: keys <= m0+15

  for (int kb = 0; kb < nkb; ++kb) {
    const int kbase = kb * 32;

    // S = Q * K^T  (two 16-key subtiles, 4 WMMA K-steps each)
    f32x8 s[2];
#pragma unroll
    for (int sub = 0; sub < 2; ++sub) {
      f32x8 st;
#pragma unroll
      for (int r = 0; r < 8; ++r) st[r] = 0.0f;
      const unsigned short* krow =
          Kb + ((size_t)kv * T + kbase + sub * 16 + l16) * HD;
#pragma unroll
      for (int j = 0; j < 4; ++j) {
        Frag kf = load_frag(krow, j * 32, half);
        st = __builtin_amdgcn_wmma_f32_16x16x32_bf16(
            false, qf[j].v, false, kf.v, (short)0, st, false, false);
      }
      s[sub] = st;
    }

    // scale + causal mask (only blocks touching the diagonal)
    const bool need_mask = (kbase + 31) > m0;
#pragma unroll
    for (int sub = 0; sub < 2; ++sub)
#pragma unroll
      for (int r = 0; r < 8; ++r) {
        float v = s[sub][r] * scale;
        if (need_mask) {
          const int qpos = m0 + r + 8 * half;
          const int kpos = kbase + sub * 16 + l16;
          v = (kpos <= qpos) ? v : -1e30f;
        }
        s[sub][r] = v;
      }

    // online softmax; rows live across the 16 lanes of a half-wave
#pragma unroll
    for (int r = 0; r < 8; ++r) {
      float mx = fmaxf(s[0][r], s[1][r]);
#pragma unroll
      for (int m = 1; m < 16; m <<= 1) mx = fmaxf(mx, __shfl_xor(mx, m, 32));
      const float newm = fmaxf(mrow[r], mx);
      const float alpha = __expf(mrow[r] - newm);
      const float p0 = __expf(s[0][r] - newm);
      const float p1 = __expf(s[1][r] - newm);
      float rs = p0 + p1;
#pragma unroll
      for (int m = 1; m < 16; m <<= 1) rs += __shfl_xor(rs, m, 32);
      lrow[r] = lrow[r] * alpha + rs;
      mrow[r] = newm;
#pragma unroll
      for (int db = 0; db < 8; ++db) o[db][r] *= alpha;
      // P in plain row-major [16 q-rows][32 keys] bf16 in LDS
      Pl[(r + 8 * half) * 32 + l16]      = f2bf(p0);
      Pl[(r + 8 * half) * 32 + 16 + l16] = f2bf(p1);
    }
    __syncthreads();   // 1-wave group: cheap, orders LDS store -> b128 load

    // reload P as an A-operand fragment (row = q, K = 32 keys)
    Frag pf;
    pf.u[0] = *(const u32x4*)(&Pl[l16 * 32 + half * 8]);
    pf.u[1] = *(const u32x4*)(&Pl[l16 * 32 + 16 + half * 8]);

    // O += P * V ; B-operand columns are rows of Vt (contiguous over keys)
#pragma unroll
    for (int db = 0; db < 8; ++db) {
      const unsigned short* vrow =
          Vt + ((size_t)kv * HD + db * 16 + l16) * T + kbase;
      Frag vf;
      vf.u[0] = *(const u32x4*)(vrow + half * 8);
      vf.u[1] = *(const u32x4*)(vrow + 16 + half * 8);
      o[db] = __builtin_amdgcn_wmma_f32_16x16x32_bf16(
          false, pf.v, false, vf.v, (short)0, o[db], false, false);
    }
    __syncthreads();
  }

  // epilogue: normalize and store [T][H*HD]
  float invl[8];
#pragma unroll
  for (int r = 0; r < 8; ++r) invl[r] = 1.0f / lrow[r];
#pragma unroll
  for (int db = 0; db < 8; ++db)
#pragma unroll
    for (int r = 0; r < 8; ++r)
      Y[(size_t)(m0 + r + 8 * half) * (N_HEADS * HD) + h * HD + db * 16 + l16] =
          o[db][r] * invl[r];
}

// ---------------- y * sigmoid(gate) -> bf16 ----------------
__global__ __launch_bounds__(256) void gate_mul(const float* __restrict__ Yf,
                                                const float* __restrict__ QG,
                                                unsigned short* __restrict__ Yg,
                                                int T) {
  const int idx = blockIdx.x * blockDim.x + threadIdx.x;   // over T * 2048
  if (idx >= T * N_EMBD) return;
  const int t  = idx >> 11;
  const int hd = idx & (N_EMBD - 1);
  const int h  = hd >> 7;
  const int d  = hd & (HD - 1);
  const float g = QG[(size_t)t * QG_N + h * 256 + 128 + d];
  const float y = Yf[idx];
  Yg[idx] = f2bf(y / (1.0f + __expf(-g)));
}

// ---------------- launcher ----------------
extern "C" void kernel_launch(void* const* d_in, const int* in_sizes, int n_in,
                              void* d_out, int out_size, void* d_ws, size_t ws_size,
                              hipStream_t stream) {
  const float* x    = (const float*)d_in[0];
  const int*   ipos = (const int*)  d_in[1];
  const float* Wq   = (const float*)d_in[2];
  const float* Wk   = (const float*)d_in[3];
  const float* Wv   = (const float*)d_in[4];
  const float* Wo   = (const float*)d_in[5];
  const float* qnw  = (const float*)d_in[6];
  const float* knw  = (const float*)d_in[7];
  float* out = (float*)d_out;

  char* ws = (char*)d_ws;
  size_t off = 0;
  auto carve = [&](size_t bytes) -> char* {
    char* p = ws + off;
    off += (bytes + 255) & ~(size_t)255;
    return p;
  };

  unsigned short* xb  = (unsigned short*)carve((size_t)T_SEQ * N_EMBD * 2);
  unsigned short* Wqb = (unsigned short*)carve((size_t)QG_N * N_EMBD * 2);
  unsigned short* Wkb = (unsigned short*)carve((size_t)KV_N * N_EMBD * 2);
  unsigned short* Wvb = (unsigned short*)carve((size_t)KV_N * N_EMBD * 2);
  unsigned short* Wob = (unsigned short*)carve((size_t)N_EMBD * N_EMBD * 2);
  float*          QGf = (float*)carve((size_t)T_SEQ * QG_N * 4);
  float*          Kpf = (float*)carve((size_t)T_SEQ * KV_N * 4);
  float*          Vpf = (float*)carve((size_t)T_SEQ * KV_N * 4);
  unsigned short* Qbf = (unsigned short*)carve((size_t)N_HEADS * T_SEQ * HD * 2);
  unsigned short* Kbf = (unsigned short*)carve((size_t)N_KV * T_SEQ * HD * 2);
  unsigned short* Vtb = (unsigned short*)carve((size_t)N_KV * HD * T_SEQ * 2);
  float*          Yf  = (float*)carve((size_t)T_SEQ * N_EMBD * 4);
  unsigned short* Ygb = (unsigned short*)carve((size_t)T_SEQ * N_EMBD * 2);

  auto cvt_grid = [](long n) { return (unsigned)((n / 4 + 255) / 256); };

  // 1) fp32 -> bf16 conversions
  cvt_f32_bf16<<<cvt_grid((long)T_SEQ * N_EMBD), 256, 0, stream>>>(x,  xb,  (long)T_SEQ * N_EMBD);
  cvt_f32_bf16<<<cvt_grid((long)QG_N * N_EMBD),  256, 0, stream>>>(Wq, Wqb, (long)QG_N * N_EMBD);
  cvt_f32_bf16<<<cvt_grid((long)KV_N * N_EMBD),  256, 0, stream>>>(Wk, Wkb, (long)KV_N * N_EMBD);
  cvt_f32_bf16<<<cvt_grid((long)KV_N * N_EMBD),  256, 0, stream>>>(Wv, Wvb, (long)KV_N * N_EMBD);
  cvt_f32_bf16<<<cvt_grid((long)N_EMBD * N_EMBD),256, 0, stream>>>(Wo, Wob, (long)N_EMBD * N_EMBD);

  // 2) projections: QG = x Wq^T, K = x Wk^T, V = x Wv^T  (TDM + WMMA)
  gemm_bf16_nt<<<dim3(QG_N / 128, T_SEQ / 128), 256, 0, stream>>>(xb, Wqb, QGf, T_SEQ, QG_N, N_EMBD);
  gemm_bf16_nt<<<dim3(KV_N / 128, T_SEQ / 128), 256, 0, stream>>>(xb, Wkb, Kpf, T_SEQ, KV_N, N_EMBD);
  gemm_bf16_nt<<<dim3(KV_N / 128, T_SEQ / 128), 256, 0, stream>>>(xb, Wvb, Vpf, T_SEQ, KV_N, N_EMBD);

  // 3) RMSNorm + RoPE -> head-major bf16 Q/K
  qk_norm_rope<<<(T_SEQ * N_HEADS) / 8, 256, 0, stream>>>(QGf, QG_N, 256, qnw, ipos, Qbf, T_SEQ, N_HEADS);
  qk_norm_rope<<<(T_SEQ * N_KV) / 8,    256, 0, stream>>>(Kpf, KV_N, HD,  knw, ipos, Kbf, T_SEQ, N_KV);

  // 4) V -> (KV*HD, T) bf16 for contiguous PV B-operand loads
  v_transpose<<<(KV_N * T_SEQ) / 256, 256, 0, stream>>>(Vpf, Vtb, T_SEQ);

  // 5) causal flash attention
  flash_attn<<<dim3(T_SEQ / 16, N_HEADS), 32, 0, stream>>>(Qbf, Kbf, Vtb, Yf, T_SEQ);

  // 6) gate
  gate_mul<<<(T_SEQ * N_EMBD) / 256, 256, 0, stream>>>(Yf, QGf, Ygb, T_SEQ);

  // 7) output projection into d_out
  gemm_bf16_nt<<<dim3(N_EMBD / 128, T_SEQ / 128), 256, 0, stream>>>(Ygb, Wob, out, T_SEQ, N_EMBD, N_EMBD);
}
